// Up2Bottom_Net_45260365365559
// MI455X (gfx1250) — compile-verified, matrix operand order checked by reading, not run on hardware
//
#include <hip/hip_runtime.h>
#include <hip/hip_bf16.h>
#include <math.h>

// ---------------- problem constants ----------------
constexpr int S_ = 4, B_ = 8, L_ = 512, D_ = 128, H_ = 128, NOP_ = 64;
constexpr int W_ = S_ * L_;          // 2048
constexpr int H3_ = 3 * H_;          // 384
constexpr float FT_ = 0.1f;

// ---------------- WMMA types ----------------
typedef __attribute__((ext_vector_type(16))) __bf16 v16bf;
typedef __attribute__((ext_vector_type(8)))  float  v8f;

#define ASTR 36   // padded LDS row stride (halfs) for the 16x32 A tile
#define CSTR 132  // padded LDS row stride (floats) for the 16x128 C slab

// A-fragment (16x32 bf16) from LDS tile.
// Layout (ISA 7.12.2): lanes 0-15 row M=lane hold K=0-7,16-23; lanes 16-31 hold K=8-15,24-31.
__device__ __forceinline__ v16bf load_a_frag(const __bf16* A_lds, int lane) {
  const int m  = lane & 15;
  const int kb = (lane >> 4) << 3;     // 0 or 8
  const __bf16* p = A_lds + m * ASTR + kb;
  v16bf a;
#pragma unroll
  for (int i = 0; i < 8; ++i) a[i] = p[i];          // K = kb + i
#pragma unroll
  for (int i = 0; i < 8; ++i) a[8 + i] = p[16 + i]; // K = 16 + kb + i
  return a;
}

// B-fragment (32x16 bf16): column N = lane&15; same K split per lane half.
// brow points at BT[n_col][k0] with row-major [N][K] storage.
__device__ __forceinline__ v16bf load_b_frag(const __bf16* brow, int lane) {
  const int kb = (lane >> 4) << 3;
  v16bf b;
#pragma unroll
  for (int i = 0; i < 8; ++i) b[i] = brow[kb + i];
#pragma unroll
  for (int i = 0; i < 8; ++i) b[8 + i] = brow[16 + kb + i];
  return b;
}

// Row-wise L2 normalization epilogue over a 16x128 slab held by 8 waves,
// writes bf16 into dst[m*dst_stride + c].
__device__ __forceinline__ void l2_epilogue(const v8f& acc, int wv, int lane, int tid,
                                            float* C_lds, float* red, float* rnorm,
                                            __bf16* dst, int dst_stride) {
  const int n  = wv * 16 + (lane & 15);
  const int mh = (lane >> 4) * 8;
#pragma unroll
  for (int r = 0; r < 8; ++r) C_lds[(mh + r) * CSTR + n] = acc[r];
  __syncthreads();
  {
    const int row = tid >> 4, g = tid & 15;
    float s = 0.f;
#pragma unroll
    for (int i = 0; i < 8; ++i) { float v = C_lds[row * CSTR + g * 8 + i]; s += v * v; }
    red[row * 16 + g] = s;
  }
  __syncthreads();
  if (tid < 16) {
    float s = 0.f;
#pragma unroll
    for (int i = 0; i < 16; ++i) s += red[tid * 16 + i];
    rnorm[tid] = 1.0f / (sqrtf(s) + 1e-30f);
  }
  __syncthreads();
  {
    const int row = tid >> 4, c0 = (tid & 15) * 8;
    const float rn = rnorm[row];
#pragma unroll
    for (int i = 0; i < 8; ++i)
      dst[(size_t)row * dst_stride + c0 + i] = (__bf16)(C_lds[row * CSTR + c0 + i] * rn);
  }
  __syncthreads();
}

// ---------------- kernel 0: transpose f32 [R][C] -> bf16 [C][R] ----------------
__global__ void k_transpose_bf16(const float* __restrict__ src, __bf16* __restrict__ dst,
                                 int R, int C) {
  int idx = blockIdx.x * blockDim.x + threadIdx.x;
  if (idx < R * C) {
    int r = idx / C, c = idx % C;
    dst[(size_t)c * R + r] = (__bf16)src[idx];
  }
}

// ---------------- kernel 1: E_T[b][h][o] = op_emb[b] @ o_w_W + o_w_b ----------------
__global__ void k_proj_ops(const float* __restrict__ op_emb,       // [B][NOP][D]
                           const __bf16* __restrict__ WT,          // [H][D]
                           const float* __restrict__ bias,         // [H]
                           __bf16* __restrict__ ET) {              // [B][H][NOP]
  const int o0 = blockIdx.x * 16, b = blockIdx.y;
  const int tid = threadIdx.x, lane = tid & 31, wv = tid >> 5;
  __shared__ __bf16 A_lds[16 * ASTR];
  v8f acc = {};
  const float* Xrow = op_emb + (((size_t)b * NOP_) + o0) * D_;
  for (int k0 = 0; k0 < D_; k0 += 32) {
    __syncthreads();
    { int e = tid * 2, r = e >> 5, c = e & 31;
      A_lds[r * ASTR + c]     = (__bf16)Xrow[(size_t)r * D_ + k0 + c];
      A_lds[r * ASTR + c + 1] = (__bf16)Xrow[(size_t)r * D_ + k0 + c + 1]; }
    __syncthreads();
    v16bf a = load_a_frag(A_lds, lane);
    v16bf bv = load_b_frag(WT + (size_t)(wv * 16 + (lane & 15)) * D_ + k0, lane);
    acc = __builtin_amdgcn_wmma_f32_16x16x32_bf16(false, a, false, bv, (short)0, acc, false, false);
  }
  const int h = wv * 16 + (lane & 15);
  const float bb = bias[h];
  const int mh = (lane >> 4) * 8;
  __bf16* dp = ET + ((size_t)b * H_ + h) * NOP_ + o0 + mh;
#pragma unroll
  for (int r = 0; r < 8; ++r) dp[r] = (__bf16)(acc[r] + bb);
}

// ---------------- kernel 2: Y_T[b][h][w] = all_word[b] @ Wmat + bias ----------------
__global__ void k_proj_words(const float* __restrict__ word_outputs, // [S][B][L][D]
                             const __bf16* __restrict__ WT,          // [H][D]
                             const float* __restrict__ bias,         // [H]
                             __bf16* __restrict__ YT) {              // [B][H][W]
  const int w0 = blockIdx.x * 16, b = blockIdx.y;
  const int tid = threadIdx.x, lane = tid & 31, wv = tid >> 5;
  __shared__ __bf16 A_lds[16 * ASTR];
  v8f acc = {};
  const int s = w0 >> 9, l0 = w0 & (L_ - 1);   // 16 rows stay inside one segment
  const float* Xrow = word_outputs + ((((size_t)s * B_) + b) * L_ + l0) * D_;
  for (int k0 = 0; k0 < D_; k0 += 32) {
    __syncthreads();
    { int e = tid * 2, r = e >> 5, c = e & 31;
      A_lds[r * ASTR + c]     = (__bf16)Xrow[(size_t)r * D_ + k0 + c];
      A_lds[r * ASTR + c + 1] = (__bf16)Xrow[(size_t)r * D_ + k0 + c + 1]; }
    __syncthreads();
    v16bf a = load_a_frag(A_lds, lane);
    v16bf bv = load_b_frag(WT + (size_t)(wv * 16 + (lane & 15)) * D_ + k0, lane);
    acc = __builtin_amdgcn_wmma_f32_16x16x32_bf16(false, a, false, bv, (short)0, acc, false, false);
  }
  const int h = wv * 16 + (lane & 15);
  const float bb = bias[h];
  const int mh = (lane >> 4) * 8;
  __bf16* dp = YT + ((size_t)b * H_ + h) * W_ + w0 + mh;
#pragma unroll
  for (int r = 0; r < 8; ++r) dp[r] = (__bf16)(acc[r] + bb);
}

// ---------------- kernel 3: word_o = l2(word_operator @ E) ----------------
__global__ void k_agg_op(const float* __restrict__ word_operator,  // [B][W][NOP]
                         const __bf16* __restrict__ ET,            // [B][H][NOP]
                         __bf16* __restrict__ neighbor) {          // [B][W][3H] sect 0
  const int v0 = blockIdx.x * 16, b = blockIdx.y;
  const int tid = threadIdx.x, lane = tid & 31, wv = tid >> 5;
  __shared__ __bf16 A_lds[16 * ASTR];
  __shared__ float C_lds[16 * CSTR];
  __shared__ float red[256];
  __shared__ float rnorm[16];
  v8f acc = {};
  const float* Arow = word_operator + ((size_t)b * W_ + v0) * NOP_;
  for (int k0 = 0; k0 < NOP_; k0 += 32) {
    __syncthreads();
    { int e = tid * 2, r = e >> 5, c = e & 31;
      A_lds[r * ASTR + c]     = (__bf16)Arow[(size_t)r * NOP_ + k0 + c];
      A_lds[r * ASTR + c + 1] = (__bf16)Arow[(size_t)r * NOP_ + k0 + c + 1]; }
    __syncthreads();
    v16bf a = load_a_frag(A_lds, lane);
    v16bf bv = load_b_frag(ET + ((size_t)b * H_ + wv * 16 + (lane & 15)) * NOP_ + k0, lane);
    acc = __builtin_amdgcn_wmma_f32_16x16x32_bf16(false, a, false, bv, (short)0, acc, false, false);
  }
  l2_epilogue(acc, wv, lane, tid, C_lds, red, rnorm,
              neighbor + ((size_t)b * W_ + v0) * H3_ + 0, H3_);
}

// ---------------- kernel 4: word_w_k = l2((ww .* we) @ Pk)  [dominant GEMM] ----------------
__global__ void k_agg_know(const float* __restrict__ ww,    // [B][W][W]
                           const float* __restrict__ we,    // [B][W][W]
                           const __bf16* __restrict__ PkT,  // [B][H][W]
                           __bf16* __restrict__ neighbor) { // sect 1 (offset H)
  const int v0 = blockIdx.x * 16, b = blockIdx.y;
  const int tid = threadIdx.x, lane = tid & 31, wv = tid >> 5;
  __shared__ __bf16 A_lds[16 * ASTR];
  __shared__ float C_lds[16 * CSTR];
  __shared__ float red[256];
  __shared__ float rnorm[16];
  v8f acc = {};
  const float* wwr = ww + ((size_t)b * W_ + v0) * W_;
  const float* wer = we + ((size_t)b * W_ + v0) * W_;
  const int e = tid * 2, ar = e >> 5, ac = e & 31;
  for (int k0 = 0; k0 < W_; k0 += 32) {
    __syncthreads();
    { const size_t o = (size_t)ar * W_ + k0 + ac;
      A_lds[ar * ASTR + ac]     = (__bf16)(wwr[o] * wer[o]);
      A_lds[ar * ASTR + ac + 1] = (__bf16)(wwr[o + 1] * wer[o + 1]);
      if (k0 + 32 < W_) {                       // prefetch next mask tile
        __builtin_prefetch(&wwr[o + 32], 0, 0);
        __builtin_prefetch(&wer[o + 32], 0, 0);
      } }
    __syncthreads();
    v16bf a = load_a_frag(A_lds, lane);
    v16bf bv = load_b_frag(PkT + ((size_t)b * H_ + wv * 16 + (lane & 15)) * W_ + k0, lane);
    acc = __builtin_amdgcn_wmma_f32_16x16x32_bf16(false, a, false, bv, (short)0, acc, false, false);
  }
  l2_epilogue(acc, wv, lane, tid, C_lds, red, rnorm,
              neighbor + ((size_t)b * W_ + v0) * H3_ + H_, H3_);
}

// ---------------- kernel 5: word_w_s = l2(depend_relation @ Ps) per segment ----------------
__global__ void k_agg_seg(const float* __restrict__ dep,   // [S][B][L][L]
                          const __bf16* __restrict__ PsT,  // [B][H][W]
                          __bf16* __restrict__ neighbor) { // sect 2 (offset 2H)
  const int l0 = blockIdx.x * 16, b = blockIdx.y, s = blockIdx.z;
  const int tid = threadIdx.x, lane = tid & 31, wv = tid >> 5;
  __shared__ __bf16 A_lds[16 * ASTR];
  __shared__ float C_lds[16 * CSTR];
  __shared__ float red[256];
  __shared__ float rnorm[16];
  v8f acc = {};
  const float* Arow = dep + ((((size_t)s * B_) + b) * L_ + l0) * L_;
  for (int k0 = 0; k0 < L_; k0 += 32) {
    __syncthreads();
    { int e = tid * 2, r = e >> 5, c = e & 31;
      A_lds[r * ASTR + c]     = (__bf16)Arow[(size_t)r * L_ + k0 + c];
      A_lds[r * ASTR + c + 1] = (__bf16)Arow[(size_t)r * L_ + k0 + c + 1]; }
    __syncthreads();
    v16bf a = load_a_frag(A_lds, lane);
    v16bf bv = load_b_frag(
        PsT + ((size_t)b * H_ + wv * 16 + (lane & 15)) * W_ + s * L_ + k0, lane);
    acc = __builtin_amdgcn_wmma_f32_16x16x32_bf16(false, a, false, bv, (short)0, acc, false, false);
  }
  l2_epilogue(acc, wv, lane, tid, C_lds, red, rnorm,
              neighbor + ((size_t)b * W_ + s * L_ + l0) * H3_ + 2 * H_, H3_);
}

// ---------------- kernel 6: word_updated^T = relu(neighbor @ up_W + up_b) ----------------
__global__ void k_update(const __bf16* __restrict__ neighbor, // [B][W][3H]
                         const __bf16* __restrict__ upWT,     // [D][3H]
                         const float* __restrict__ up_b,      // [D]
                         float* __restrict__ WUT) {           // [B][D][W]
  const int w0 = blockIdx.x * 16, b = blockIdx.y;
  const int tid = threadIdx.x, lane = tid & 31, wv = tid >> 5;
  __shared__ __bf16 A_lds[16 * ASTR];
  v8f acc = {};
  const __bf16* Arow = neighbor + ((size_t)b * W_ + w0) * H3_;
  for (int k0 = 0; k0 < H3_; k0 += 32) {
    __syncthreads();
    { int e = tid * 2, r = e >> 5, c = e & 31;
      A_lds[r * ASTR + c]     = Arow[(size_t)r * H3_ + k0 + c];
      A_lds[r * ASTR + c + 1] = Arow[(size_t)r * H3_ + k0 + c + 1]; }
    __syncthreads();
    v16bf a = load_a_frag(A_lds, lane);
    v16bf bv = load_b_frag(upWT + (size_t)(wv * 16 + (lane & 15)) * H3_ + k0, lane);
    acc = __builtin_amdgcn_wmma_f32_16x16x32_bf16(false, a, false, bv, (short)0, acc, false, false);
  }
  const int d = wv * 16 + (lane & 15);
  const float bb = up_b[d];
  const int mh = (lane >> 4) * 8;
  float* dp = WUT + ((size_t)b * D_ + d) * W_ + w0 + mh;
#pragma unroll
  for (int r = 0; r < 8; ++r) { float v = acc[r] + bb; dp[r] = v > 0.f ? v : 0.f; }
}

// ---------------- kernel 7: goal pooling + gates + output ----------------
__global__ void k_final(const float* __restrict__ goal_word, const float* __restrict__ ex_seq,
                        const float* __restrict__ WUT,        // [B][D][W]
                        const float* __restrict__ node_hidden,// [B][D]
                        const float* __restrict__ wg_W, const float* __restrict__ wg_b,
                        const float* __restrict__ fg_W, const float* __restrict__ fg_b,
                        float* __restrict__ out) {            // [B][D]
  const int b = blockIdx.x, tid = threadIdx.x; // 256 threads
  __shared__ float msk[W_];
  __shared__ float red[256];
  __shared__ float gw[D_];
  float cnt = 0.f;
  for (int w = tid; w < W_; w += 256) {
    float v = goal_word[(size_t)b * W_ + w] * ex_seq[(size_t)b * W_ + w];
    msk[w] = v;
    cnt += (v != 0.f) ? 1.f : 0.f;
  }
  red[tid] = cnt;
  __syncthreads();
  for (int st = 128; st > 0; st >>= 1) { if (tid < st) red[tid] += red[tid + st]; __syncthreads(); }
  const float inv = 1.f / (red[0] + 1e-30f);
  __syncthreads();
  const int d = tid & 127, half = tid >> 7;
  const float* wrow = WUT + ((size_t)b * D_ + d) * W_ + half * (W_ / 2);
  const float* mr = msk + half * (W_ / 2);
  float s = 0.f;
  for (int w = 0; w < W_ / 2; ++w) s += mr[w] * wrow[w];
  red[tid] = s;
  __syncthreads();
  if (tid < D_) gw[tid] = (red[tid] + red[tid + 128]) * inv;
  __syncthreads();
  if (tid < D_) {
    float gu = wg_b[d];
    for (int k = 0; k < D_; ++k) gu += gw[k] * wg_W[(size_t)k * D_ + d];
    gu = gu > 0.f ? gu : 0.f;
    float f = fg_b[d];
    for (int k = 0; k < D_; ++k) f += gw[k] * fg_W[(size_t)k * D_ + d];
    for (int k = 0; k < D_; ++k) f += node_hidden[(size_t)b * D_ + k] * fg_W[(size_t)(D_ + k) * D_ + d];
    f = 1.f / (1.f + expf(-f));
    const float ff = f > FT_ ? f : FT_;
    out[(size_t)b * D_ + d] = ff * node_hidden[(size_t)b * D_ + d] + (1.f - f) * gu;
  }
}

// ---------------- host launcher ----------------
extern "C" void kernel_launch(void* const* d_in, const int* in_sizes, int n_in,
                              void* d_out, int out_size, void* d_ws, size_t ws_size,
                              hipStream_t stream) {
  const float* word_outputs = (const float*)d_in[0];
  const float* node_hidden  = (const float*)d_in[1];
  const float* op_embedding = (const float*)d_in[2];
  const float* word_operator= (const float*)d_in[3];
  const float* word_word    = (const float*)d_in[4];
  const float* depend_rel   = (const float*)d_in[5];
  const float* word_exist_m = (const float*)d_in[6];
  const float* word_exist_s = (const float*)d_in[7];
  const float* goal_word    = (const float*)d_in[8];
  const float* o_w_W = (const float*)d_in[9];  const float* o_w_b = (const float*)d_in[10];
  const float* wk_W  = (const float*)d_in[11]; const float* wk_b  = (const float*)d_in[12];
  const float* ws_W  = (const float*)d_in[13]; const float* ws_b  = (const float*)d_in[14];
  const float* up_W  = (const float*)d_in[15]; const float* up_b  = (const float*)d_in[16];
  const float* wg_W  = (const float*)d_in[17]; const float* wg_b  = (const float*)d_in[18];
  const float* fg_W  = (const float*)d_in[19]; const float* fg_b  = (const float*)d_in[20];
  float* out = (float*)d_out;

  // workspace carve-up (256B aligned blocks)
  char* ws = (char*)d_ws;
  size_t off = 0;
  auto alloc = [&](size_t bytes) { char* p = ws + off; off = (off + bytes + 255) & ~size_t(255); return p; };
  __bf16* oWT  = (__bf16*)alloc((size_t)H_ * D_ * 2);      // [H][D]
  __bf16* wkWT = (__bf16*)alloc((size_t)H_ * D_ * 2);      // [H][D]
  __bf16* wsWT = (__bf16*)alloc((size_t)H_ * D_ * 2);      // [H][D]
  __bf16* upWT = (__bf16*)alloc((size_t)D_ * H3_ * 2);     // [D][3H]
  __bf16* ET   = (__bf16*)alloc((size_t)B_ * H_ * NOP_ * 2);// [B][H][NOP]
  __bf16* PkT  = (__bf16*)alloc((size_t)B_ * H_ * W_ * 2); // [B][H][W]
  __bf16* PsT  = (__bf16*)alloc((size_t)B_ * H_ * W_ * 2); // [B][H][W]
  __bf16* nbr  = (__bf16*)alloc((size_t)B_ * W_ * H3_ * 2);// [B][W][3H]
  float*  WUT  = (float*) alloc((size_t)B_ * D_ * W_ * 4); // [B][D][W]
  (void)ws_size; (void)in_sizes; (void)n_in; (void)out_size;

  // 0) weight transposes to bf16
  k_transpose_bf16<<<(D_ * H_ + 255) / 256, 256, 0, stream>>>(o_w_W, oWT, D_, H_);
  k_transpose_bf16<<<(D_ * H_ + 255) / 256, 256, 0, stream>>>(wk_W, wkWT, D_, H_);
  k_transpose_bf16<<<(D_ * H_ + 255) / 256, 256, 0, stream>>>(ws_W, wsWT, D_, H_);
  k_transpose_bf16<<<(H3_ * D_ + 255) / 256, 256, 0, stream>>>(up_W, upWT, H3_, D_);

  // 1) projections
  k_proj_ops  <<<dim3(NOP_ / 16, B_), 256, 0, stream>>>(op_embedding, oWT, o_w_b, ET);
  k_proj_words<<<dim3(W_ / 16, B_),   256, 0, stream>>>(word_outputs, wkWT, wk_b, PkT);
  k_proj_words<<<dim3(W_ / 16, B_),   256, 0, stream>>>(word_outputs, wsWT, ws_b, PsT);

  // 2) aggregations (L2-normalized; _norm_mat row scaling cancels under L2)
  k_agg_op  <<<dim3(W_ / 16, B_),      256, 0, stream>>>(word_operator, ET, nbr);
  k_agg_know<<<dim3(W_ / 16, B_),      256, 0, stream>>>(word_word, word_exist_m, PkT, nbr);
  k_agg_seg <<<dim3(L_ / 16, B_, S_),  256, 0, stream>>>(depend_rel, PsT, nbr);

  // 3) up-projection
  k_update<<<dim3(W_ / 16, B_), 256, 0, stream>>>(nbr, upWT, up_b, WUT);

  // 4) goal pooling + gates
  k_final<<<B_, 256, 0, stream>>>(goal_word, word_exist_s, WUT, node_hidden,
                                  wg_W, wg_b, fg_W, fg_b, out);
}
